// Mamba_71150428225871
// MI455X (gfx1250) — compile-verified, matrix operand order checked by reading, not run on hardware
//
#include <hip/hip_runtime.h>
#include <hip/hip_bf16.h>
#include <math.h>

// ---- problem dims (compile-time) ----
#define BB   4
#define LL   1024
#define INF  256
#define DMV  1024
#define DSV  16
#define DCV  4
#define DIV  2048          // d_inner
#define DTRV 64            // dt_rank
#define NXP  96            // DTR + 2*DS
#define MM   (BB*LL)       // 4096 rows

typedef __bf16 bf16_t;
typedef __attribute__((ext_vector_type(16))) __bf16 v16bf;
typedef __attribute__((ext_vector_type(8)))  __bf16 v8bf;
typedef __attribute__((ext_vector_type(8)))  float  v8f;

// ---------------------------------------------------------------------------
// elementwise f32 -> bf16 convert
// ---------------------------------------------------------------------------
__global__ void cvt_f32_bf16(const float* __restrict__ src, bf16_t* __restrict__ dst, int n) {
    int i = blockIdx.x * blockDim.x + threadIdx.x;
    if (i < n) dst[i] = (bf16_t)src[i];
}

// ---------------------------------------------------------------------------
// WMMA bf16 GEMM:  C[M,N] = A[M,K] @ W[N,K]^T  (+bias[n])
// A row-major (lda), W row-major (ldw) -> B-fragment (col n of W^T) is a
// contiguous strip of W row n.  Each wave owns a 16 x 32 strip (NT=2): the A
// fragment is reused across both N-tiles.  K-steps are MANUALLY DOUBLE-
// BUFFERED: fragments for step k+32 are loaded into a fresh register set
// before the WMMAs of step k issue, so the WMMAs depend only on loads issued
// a full iteration earlier (partial s_wait_loadcnt, in-wave overlap) even
// under a pressure-minimizing scheduler.
// Per-lane 16-bit fragment layout (ISA 7.12.2):
//   elems 0..7  -> k = k0 + (lane>>4)*8 + 0..7
//   elems 8..15 -> k = k0 + 16 + (lane>>4)*8 + 0..7
// C/D: lane l holds col n=(l&15), rows m = r + 8*(l>>4), r=0..7.
// ---------------------------------------------------------------------------
#define NT 2
__global__ __launch_bounds__(256)
void gemm_wmma_bf16(const bf16_t* __restrict__ A, int lda,
                    const bf16_t* __restrict__ W, int ldw,
                    float* __restrict__ Cf, bf16_t* __restrict__ Cb, int ldc,
                    const float* __restrict__ bias,
                    int M, int N, int K)
{
    const int wave = threadIdx.x >> 5;
    const int lane = threadIdx.x & 31;
    const int stripsN = N / (16 * NT);
    const int totalStrips = (M >> 4) * stripsN;
    const int stripId = blockIdx.x * (blockDim.x >> 5) + wave;  // wave-uniform
    if (stripId >= totalStrips) return;                         // uniform exit

    const int tM = stripId / stripsN;
    const int tN = stripId - tM * stripsN;
    const int khalf = (lane >> 4) << 3;                         // 0 or 8

    const bf16_t* pA = A + (size_t)(tM * 16 + (lane & 15)) * lda + khalf;
    const bf16_t* pW = W + (size_t)(tN * (16 * NT) + (lane & 15)) * ldw + khalf;

    v8f acc[NT];
#pragma unroll
    for (int j = 0; j < NT; ++j) acc[j] = (v8f){};

    // ---- prologue: load K-step 0 (K >= 32 always holds here) ----
    v8bf a0c = *(const v8bf*)(pA);
    v8bf a1c = *(const v8bf*)(pA + 16);
    v8bf b0c[NT], b1c[NT];
#pragma unroll
    for (int j = 0; j < NT; ++j) {
        const bf16_t* pWj = pW + (size_t)j * 16 * ldw;
        b0c[j] = *(const v8bf*)(pWj);
        b1c[j] = *(const v8bf*)(pWj + 16);
    }

    for (int k = 32; k < K; k += 32) {
        // ---- load next K-step into fresh registers (no dep from WMMAs) ----
        v8bf a0n = *(const v8bf*)(pA + k);
        v8bf a1n = *(const v8bf*)(pA + k + 16);
        v8bf b0n[NT], b1n[NT];
#pragma unroll
        for (int j = 0; j < NT; ++j) {
            const bf16_t* pWj = pW + (size_t)j * 16 * ldw + k;
            b0n[j] = *(const v8bf*)(pWj);
            b1n[j] = *(const v8bf*)(pWj + 16);
        }
        // ---- compute on current buffers (loads from previous iteration) ----
        v16bf av;
#pragma unroll
        for (int i = 0; i < 8; ++i) { av[i] = a0c[i]; av[i + 8] = a1c[i]; }
#pragma unroll
        for (int j = 0; j < NT; ++j) {
            v16bf bv;
#pragma unroll
            for (int i = 0; i < 8; ++i) { bv[i] = b0c[j][i]; bv[i + 8] = b1c[j][i]; }
            acc[j] = __builtin_amdgcn_wmma_f32_16x16x32_bf16(
                         false, av, false, bv, (short)0, acc[j], false, false);
        }
        // ---- rotate (eliminated by unroll-by-2 role swap) ----
        a0c = a0n; a1c = a1n;
#pragma unroll
        for (int j = 0; j < NT; ++j) { b0c[j] = b0n[j]; b1c[j] = b1n[j]; }
    }

    // ---- epilogue: last K-step ----
    {
        v16bf av;
#pragma unroll
        for (int i = 0; i < 8; ++i) { av[i] = a0c[i]; av[i + 8] = a1c[i]; }
#pragma unroll
        for (int j = 0; j < NT; ++j) {
            v16bf bv;
#pragma unroll
            for (int i = 0; i < 8; ++i) { bv[i] = b0c[j][i]; bv[i + 8] = b1c[j][i]; }
            acc[j] = __builtin_amdgcn_wmma_f32_16x16x32_bf16(
                         false, av, false, bv, (short)0, acc[j], false, false);
        }
    }

    const int r0 = tM * 16 + ((lane >> 4) << 3);
#pragma unroll
    for (int j = 0; j < NT; ++j) {
        const int cn = tN * (16 * NT) + j * 16 + (lane & 15);
        const float bval = bias ? bias[cn] : 0.0f;
#pragma unroll
        for (int r = 0; r < 8; ++r) {
            float v = acc[j][r] + bval;
            size_t idx = (size_t)(r0 + r) * ldc + cn;
            if (Cf) Cf[idx] = v;
            if (Cb) Cb[idx] = (bf16_t)v;
        }
    }
}

// ---------------------------------------------------------------------------
// causal depthwise conv (DC=4) + bias + SiLU ;  u_pre [B,L,DI] f32
// ---------------------------------------------------------------------------
__global__ void conv_silu(const float* __restrict__ u_pre,
                          const float* __restrict__ conv_w,   // [DI,1,DC]
                          const float* __restrict__ conv_b,   // [DI]
                          float* __restrict__ u_post, bf16_t* __restrict__ u_post_bf)
{
    int idx = blockIdx.x * blockDim.x + threadIdx.x;
    if (idx >= BB * LL * DIV) return;
    int d = idx % DIV;
    int t = (idx / DIV) % LL;
    int b = idx / (DIV * LL);
    float acc = conv_b[d];
#pragma unroll
    for (int j = 0; j < DCV; ++j) {
        int tt = t - (DCV - 1) + j;
        if (tt >= 0)
            acc += conv_w[d * DCV + j] * u_pre[((size_t)(b * LL + tt)) * DIV + d];
    }
    float s = acc / (1.0f + expf(-acc));        // silu
    u_post[idx] = s;
    u_post_bf[idx] = (bf16_t)s;
}

// ---------------------------------------------------------------------------
// selective scan: one thread per (b,d) channel, DS=16 states in registers.
// B_t/C_t (xdbl cols 64..95) staged in LDS 32 timesteps at a time.
// Only the last timestep's y is needed; adds D*u skip.
// ---------------------------------------------------------------------------
__global__ __launch_bounds__(256)
void scan_kernel(const float* __restrict__ xdbl,       // [M, 96] f32
                 const float* __restrict__ delta_pre,  // [M, DI] f32 (pre-bias/softplus)
                 const float* __restrict__ u_post,     // [M, DI] f32
                 const float* __restrict__ A_log,      // [DI, DS]
                 const float* __restrict__ dt_bias,    // [DI]
                 const float* __restrict__ Dvec,       // [DI]
                 float* __restrict__ yfin)             // [B, DI]
{
    const int b = blockIdx.x;
    const int d = blockIdx.y * 256 + threadIdx.x;

    float As[DSV], hs[DSV];
#pragma unroll
    for (int s = 0; s < DSV; ++s) { As[s] = -expf(A_log[d * DSV + s]); hs[s] = 0.0f; }
    const float dtb = dt_bias[d];
    const float Dd  = Dvec[d];

    __shared__ float sB[32][DSV];
    __shared__ float sC[32][DSV];

    float y = 0.0f, ulast = 0.0f;
    for (int t0 = 0; t0 < LL; t0 += 32) {
        __syncthreads();
        for (int i = threadIdx.x; i < 32 * 32; i += 256) {
            int tt = i >> 5, c = i & 31;
            float v = xdbl[(size_t)(b * LL + t0 + tt) * NXP + DTRV + c];
            if (c < DSV) sB[tt][c] = v; else sC[tt][c - DSV] = v;
        }
        __syncthreads();
#pragma unroll 4
        for (int ti = 0; ti < 32; ++ti) {
            const size_t row = (size_t)(b * LL + t0 + ti);
            float dp = delta_pre[row * DIV + d] + dtb;
            float dv = (dp > 20.0f) ? dp : log1pf(expf(dp));   // softplus
            float uv = u_post[row * DIV + d];
            float du = dv * uv;
#pragma unroll
            for (int s = 0; s < DSV; ++s)
                hs[s] = __expf(dv * As[s]) * hs[s] + du * sB[ti][s];
            if (t0 + ti == LL - 1) {
                float acc = 0.0f;
#pragma unroll
                for (int s = 0; s < DSV; ++s) acc += hs[s] * sC[ti][s];
                y = acc; ulast = uv;
            }
        }
    }
    yfin[b * DIV + d] = y + Dd * ulast;
}

// ---------------------------------------------------------------------------
// z gate at last token only:  z = h[b,L-1,:] @ in_proj_w[DI+j,:]^T ;
// out_y = yfin * silu(z)
// ---------------------------------------------------------------------------
__global__ void zlast_gate(const bf16_t* __restrict__ h_bf,     // [M, DM]
                           const float* __restrict__ in_proj_w, // [2*DI, DM]
                           const float* __restrict__ yfin,      // [B, DI]
                           float* __restrict__ out_y)           // [B, DI]
{
    int idx = blockIdx.x * blockDim.x + threadIdx.x;
    if (idx >= BB * DIV) return;
    int b = idx / DIV, j = idx % DIV;
    const bf16_t* hrow = h_bf + (size_t)(b * LL + (LL - 1)) * DMV;
    const float*  wrow = in_proj_w + (size_t)(DIV + j) * DMV;
    float acc = 0.0f;
    for (int k = 0; k < DMV; ++k) acc += (float)hrow[k] * wrow[k];
    float sil = acc / (1.0f + expf(-acc));
    out_y[idx] = yfin[idx] * sil;
}

// out_last[b,m] = out_y[b,:] @ out_proj_w[m,:]
__global__ void out_proj_kernel(const float* __restrict__ out_y,
                                const float* __restrict__ out_proj_w, // [DM, DI]
                                float* __restrict__ out_last)         // [B, DM]
{
    int idx = blockIdx.x * blockDim.x + threadIdx.x;
    if (idx >= BB * DMV) return;
    int b = idx / DMV, m = idx % DMV;
    const float* yrow = out_y + (size_t)b * DIV;
    const float* wrow = out_proj_w + (size_t)m * DIV;
    float acc = 0.0f;
    for (int k = 0; k < DIV; ++k) acc += yrow[k] * wrow[k];
    out_last[idx] = acc;
}

// out[b] = out_last[b,:] @ fc_w + fc_b
__global__ void fc_kernel(const float* __restrict__ out_last,
                          const float* __restrict__ fc_w, const float* __restrict__ fc_b,
                          float* __restrict__ out)
{
    __shared__ float red[256];
    int b = blockIdx.x;
    float acc = 0.0f;
    for (int k = threadIdx.x; k < DMV; k += 256)
        acc += out_last[(size_t)b * DMV + k] * fc_w[k];
    red[threadIdx.x] = acc;
    __syncthreads();
    for (int s = 128; s > 0; s >>= 1) {
        if (threadIdx.x < s) red[threadIdx.x] += red[threadIdx.x + s];
        __syncthreads();
    }
    if (threadIdx.x == 0) out[b] = red[0] + fc_b[0];
}

// ---------------------------------------------------------------------------
extern "C" void kernel_launch(void* const* d_in, const int* in_sizes, int n_in,
                              void* d_out, int out_size, void* d_ws, size_t ws_size,
                              hipStream_t stream) {
    const float* x         = (const float*)d_in[0];   // [B,L,IN]
    const float* W_emb     = (const float*)d_in[1];   // [DM,IN]
    const float* b_emb     = (const float*)d_in[2];   // [DM]
    const float* in_proj_w = (const float*)d_in[3];   // [2*DI,DM]
    const float* conv_w    = (const float*)d_in[4];   // [DI,1,DC]
    const float* conv_b    = (const float*)d_in[5];   // [DI]
    const float* x_proj_w  = (const float*)d_in[6];   // [NXP,DI]
    const float* dt_proj_w = (const float*)d_in[7];   // [DI,DTR]
    const float* dt_proj_b = (const float*)d_in[8];   // [DI]
    const float* A_log     = (const float*)d_in[9];   // [DI,DS]
    const float* Dvec      = (const float*)d_in[10];  // [DI]
    const float* out_proj_w= (const float*)d_in[11];  // [DM,DI]
    const float* fc_w      = (const float*)d_in[12];  // [1,DM]
    const float* fc_b      = (const float*)d_in[13];  // [1]
    float* out = (float*)d_out;                       // [B,1]

    // ---- workspace carve-out ----
    char* ws = (char*)d_ws;
    auto carve = [&](size_t bytes) -> void* {
        void* p = ws; ws += (bytes + 255) & ~(size_t)255; return p;
    };
    bf16_t* x_bf      = (bf16_t*)carve((size_t)MM * INF * 2);
    bf16_t* wemb_bf   = (bf16_t*)carve((size_t)DMV * INF * 2);
    bf16_t* h_bf      = (bf16_t*)carve((size_t)MM * DMV * 2);
    bf16_t* wu_bf     = (bf16_t*)carve((size_t)DIV * DMV * 2);
    float*  u_pre     = (float*) carve((size_t)MM * DIV * 4);
    float*  u_post    = (float*) carve((size_t)MM * DIV * 4);
    bf16_t* u_post_bf = (bf16_t*)carve((size_t)MM * DIV * 2);
    bf16_t* xproj_bf  = (bf16_t*)carve((size_t)NXP * DIV * 2);
    float*  xdbl_f    = (float*) carve((size_t)MM * NXP * 4);
    bf16_t* xdbl_bf   = (bf16_t*)carve((size_t)MM * NXP * 2);
    bf16_t* dtproj_bf = (bf16_t*)carve((size_t)DIV * DTRV * 2);
    float*  delta_pre = (float*) carve((size_t)MM * DIV * 4);
    float*  yfin      = (float*) carve((size_t)BB * DIV * 4);
    float*  out_y     = (float*) carve((size_t)BB * DIV * 4);
    float*  out_last  = (float*) carve((size_t)BB * DMV * 4);

    auto cvt = [&](const float* s, bf16_t* d, int n) {
        cvt_f32_bf16<<<(n + 255) / 256, 256, 0, stream>>>(s, d, n);
    };
    cvt(x,         x_bf,      MM * INF);
    cvt(W_emb,     wemb_bf,   DMV * INF);
    cvt(in_proj_w, wu_bf,     DIV * DMV);     // first DI rows = u weights
    cvt(x_proj_w,  xproj_bf,  NXP * DIV);
    cvt(dt_proj_w, dtproj_bf, DIV * DTRV);

    // NT=2 strip GEMM (N % 32 == 0 for all GEMMs in this net)
    auto gemm = [&](const bf16_t* A, int lda, const bf16_t* W, int ldw,
                    float* Cf, bf16_t* Cb, int ldc, const float* bias,
                    int M, int N, int K) {
        int strips = (M / 16) * (N / 32);
        int blocks = (strips + 7) / 8;                // 8 waves / block
        gemm_wmma_bf16<<<blocks, 256, 0, stream>>>(A, lda, W, ldw, Cf, Cb, ldc,
                                                   bias, M, N, K);
    };

    // 1) h = x @ W_emb^T + b_emb           -> bf16
    gemm(x_bf, INF, wemb_bf, INF, nullptr, h_bf, DMV, b_emb, MM, DMV, INF);
    // 2) u_pre = h @ in_proj_w[:DI]^T      -> f32
    gemm(h_bf, DMV, wu_bf, DMV, u_pre, nullptr, DIV, nullptr, MM, DIV, DMV);
    // 3) causal depthwise conv + SiLU
    {
        int n = BB * LL * DIV;
        conv_silu<<<(n + 255) / 256, 256, 0, stream>>>(u_pre, conv_w, conv_b,
                                                       u_post, u_post_bf);
    }
    // 4) xdbl = u @ x_proj_w^T             -> f32 (+bf16 for dt slice); N=96
    gemm(u_post_bf, DIV, xproj_bf, DIV, xdbl_f, xdbl_bf, NXP, nullptr, MM, NXP, DIV);
    // 5) delta_pre = dt @ dt_proj_w^T      -> f32  (A slice: lda=NXP, cols 0..63)
    gemm(xdbl_bf, NXP, dtproj_bf, DTRV, delta_pre, nullptr, DIV, nullptr, MM, DIV, DTRV);
    // 6) selective scan (last-token y only, with D*u skip)
    {
        dim3 g(BB, DIV / 256);
        scan_kernel<<<g, 256, 0, stream>>>(xdbl_f, delta_pre, u_post,
                                           A_log, dt_proj_b, Dvec, yfin);
    }
    // 7) z gate at last token + multiply
    {
        int n = BB * DIV;
        zlast_gate<<<(n + 255) / 256, 256, 0, stream>>>(h_bf, in_proj_w, yfin, out_y);
    }
    // 8) out_proj at last token
    {
        int n = BB * DMV;
        out_proj_kernel<<<(n + 255) / 256, 256, 0, stream>>>(out_y, out_proj_w, out_last);
    }
    // 9) fc head
    fc_kernel<<<BB, 256, 0, stream>>>(out_last, fc_w, fc_b, out);
}
